// SinkhornAlign_54812372632137
// MI455X (gfx1250) — compile-verified, matrix operand order checked by reading, not run on hardware
//
#include <hip/hip_runtime.h>
#include <cmath>

// ---------------------------------------------------------------------------
// Sinkhorn divergence, fused GEMM (WMMA f16) + online logsumexp softmin.
// B=4, N=M=2048, D=512. Cost C = 1 - <x,y> for unit-normalized x,y.
// ---------------------------------------------------------------------------

#define BB 4
#define NN 2048
#define DD 512

typedef __attribute__((ext_vector_type(16))) _Float16 v16h;
typedef __attribute__((ext_vector_type(8)))  _Float16 v8h;
typedef __attribute__((ext_vector_type(8)))  float    v8f;

__device__ __forceinline__ v16h load_frag(const _Float16* __restrict__ rowptr,
                                          int kbase, int g) {
  // 16-bit A/B fragment layout (ISA 7.12.2): lane (g = lane>>4) holds
  // K = kbase + g*8 + 0..7 in halves 0..7, K = kbase+16 + g*8 + 0..7 in 8..15
  union { v16h v; v8h h[2]; } u;
  u.h[0] = *(const v8h*)(rowptr + kbase + g * 8);
  u.h[1] = *(const v8h*)(rowptr + kbase + 16 + g * 8);
  return u.v;
}

// --------------------------- normalize f32 -> f16 --------------------------
__global__ __launch_bounds__(128) void sk_norm_kernel(
    const float* __restrict__ in, _Float16* __restrict__ out, int nrows) {
  const int row  = blockIdx.x * 4 + (threadIdx.x >> 5);
  const int lane = threadIdx.x & 31;
  if (row >= nrows) return;
  const float* src = in + (size_t)row * DD + lane * 16;
  float vals[16];
  float ss = 0.f;
#pragma unroll
  for (int j = 0; j < 16; ++j) { vals[j] = src[j]; ss += vals[j] * vals[j]; }
#pragma unroll
  for (int off = 1; off < 32; off <<= 1) ss += __shfl_xor(ss, off, 32);
  const float scale = 1.0f / fmaxf(sqrtf(ss), 1e-12f);
  _Float16* dst = out + (size_t)row * DD + lane * 16;
#pragma unroll
  for (int j = 0; j < 16; ++j) dst[j] = (_Float16)(vals[j] * scale);
}

// ------------------- fused softmin (GEMM + online LSE) ----------------------
// mode: 0 = init  (h = logw;           out = softmin)
//       1 = body  (h = logw + pot/eps; out = 0.5*(pot_self + softmin))
//       2 = final (h = logw + pot/eps; out = softmin)
// potentials packed as [q][B][N], q: 0=f_ba 1=g_ab 2=f_aa 3=g_bb
__global__ __launch_bounds__(128) void sk_softmin_kernel(
    const _Float16* __restrict__ Xh, const _Float16* __restrict__ Yh,
    const float* __restrict__ pots_in, float* __restrict__ pots_out,
    float eps, int mode) {
  const int p    = blockIdx.z;       // which of the 4 softmin problems
  const int b    = blockIdx.y;
  const int wave = threadIdx.x >> 5;
  const int lane = threadIdx.x & 31;
  const int g    = lane >> 4;
  const int lcol = lane & 15;
  const int rowbase = (blockIdx.x * 4 + wave) * 16;

  // p=0: ft_ba = softmin(C_xy, g_ab) ; p=1: gt_ab = softmin(C_yx, f_ba)
  // p=2: ft_aa = softmin(C_xx, f_aa) ; p=3: gt_bb = softmin(C_yy, g_bb)
  const _Float16* X = (p & 1) ? Yh : Xh;
  const _Float16* Y = (p == 0 || p == 3) ? Yh : Xh;
  const int qin  = (p == 0) ? 1 : (p == 1) ? 0 : p;
  const int qout = p;
  X += (size_t)b * NN * DD;
  Y += (size_t)b * NN * DD;
  const float* potin   = pots_in + ((size_t)qin  * BB + b) * NN;
  const float* potself = pots_in + ((size_t)qout * BB + b) * NN;
  float*       outp    = pots_out + ((size_t)qout * BB + b) * NN;

  const float inv_eps = 1.0f / eps;
  const float L2E     = 1.44269504089f;     // log2(e)
  const float LN2     = 0.69314718056f;
  const float ieps2   = inv_eps * L2E;      // scale into log2 domain
  const float LOGW    = -7.6246190f;        // -log(2048), N == M

  // A fragments for this wave's 16 rows: 16 k-chunks of 32 (D = 512)
  v16h afrag[16];
  const _Float16* arow = X + (size_t)(rowbase + lcol) * DD;
#pragma unroll
  for (int kc = 0; kc < 16; ++kc) afrag[kc] = load_frag(arow, kc * 32, g);

  // running (max, sum) per owned row, in log2 domain
  float rmax[8], rsum[8];
#pragma unroll
  for (int i = 0; i < 8; ++i) { rmax[i] = -3.0e38f; rsum[i] = 0.f; }

  // --- software-pipelined m loop: 4-slot B buffer, loads run 2 chunks ahead
  const int last = NN / 16 - 1;
  v16h bq[4];
  {
    const _Float16* yrow0 = Y + (size_t)lcol * DD;
    bq[0] = load_frag(yrow0, 0, g);
    bq[1] = load_frag(yrow0, 32, g);
  }

  for (int mt = 0; mt <= last; ++mt) {
    const _Float16* yrow  = Y + (size_t)(mt * 16 + lcol) * DD;
    const int       mtn   = (mt == last) ? 0 : (mt + 1);
    const _Float16* yrown = Y + (size_t)(mtn * 16 + lcol) * DD;
    __builtin_prefetch((const void*)yrown, 0, 3);

    v8f acc = {0.f, 0.f, 0.f, 0.f, 0.f, 0.f, 0.f, 0.f};
#pragma unroll
    for (int kc = 0; kc < 16; ++kc) {
      const int c2 = kc + 2;                       // chunk to fetch ahead
      const _Float16* lp = (c2 < 16) ? yrow : yrown;
      bq[c2 & 3] = load_frag(lp, (c2 & 15) * 32, g);
      acc = __builtin_amdgcn_wmma_f32_16x16x32_f16(
          /*neg_a=*/false, afrag[kc], /*neg_b=*/false, bq[kc & 3],
          /*c_mod=*/(short)0, acc, /*reuse_a=*/false, /*reuse_b=*/false);
    }

    float pv = 0.f;
    if (mode != 0) pv = potin[mt * 16 + lcol];
    const float shift = pv - 1.0f;   // u = (G - 1 + pot)/eps  (+logw at end)
#pragma unroll
    for (int i = 0; i < 8; ++i) {
      const float u2 = (acc[i] + shift) * ieps2;   // log2-domain exponent
      const float nm = fmaxf(rmax[i], u2);
      rsum[i] = rsum[i] * exp2f(rmax[i] - nm) + exp2f(u2 - nm);
      rmax[i] = nm;
    }
  }

  // reduce (max,sum) across the 16 lanes of each half-wave
#pragma unroll
  for (int off = 1; off < 16; off <<= 1) {
#pragma unroll
    for (int i = 0; i < 8; ++i) {
      const float om = __shfl_xor(rmax[i], off, 32);
      const float os = __shfl_xor(rsum[i], off, 32);
      const float nm = fmaxf(rmax[i], om);
      rsum[i] = rsum[i] * exp2f(rmax[i] - nm) + os * exp2f(om - nm);
      rmax[i] = nm;
    }
  }

  if (lcol == 0) {
    const int rb = rowbase + 8 * g;  // lane0 -> rows 0..7, lane16 -> rows 8..15
#pragma unroll
    for (int i = 0; i < 8; ++i) {
      const float sm = -eps * (LOGW + LN2 * (rmax[i] + log2f(rsum[i])));
      const float o  = (mode == 1) ? 0.5f * (potself[rb + i] + sm) : sm;
      outp[rb + i] = o;
    }
  }
}

// ------------------------------ final loss ----------------------------------
__global__ __launch_bounds__(256) void sk_loss_kernel(
    const float* __restrict__ finals, float* __restrict__ out) {
  __shared__ float red[256];
  const int T = BB * NN;
  float s = 0.f;
  for (int idx = threadIdx.x; idx < T; idx += blockDim.x)
    s += finals[0 * T + idx] - finals[2 * T + idx] +
         finals[1 * T + idx] - finals[3 * T + idx];
  red[threadIdx.x] = s;
  __syncthreads();
  for (int off = 128; off > 0; off >>= 1) {
    if ((int)threadIdx.x < off) red[threadIdx.x] += red[threadIdx.x + off];
    __syncthreads();
  }
  if (threadIdx.x == 0) out[0] = red[0] / (float)(BB * NN);
}

// ------------------------------ launcher ------------------------------------
extern "C" void kernel_launch(void* const* d_in, const int* in_sizes, int n_in,
                              void* d_out, int out_size, void* d_ws,
                              size_t ws_size, hipStream_t stream) {
  (void)in_sizes; (void)n_in; (void)out_size; (void)ws_size;
  const float* Ep = (const float*)d_in[0];   // (B, N, D) f32
  const float* Et = (const float*)d_in[1];   // (B, M, D) f32
  float* out = (float*)d_out;                // scalar f32

  // workspace layout
  const size_t XBYTES = (size_t)BB * NN * DD * sizeof(_Float16);  // 8 MB
  char* ws = (char*)d_ws;
  _Float16* Xh    = (_Float16*)ws;
  _Float16* Yh    = (_Float16*)(ws + XBYTES);
  float*    pots0 = (float*)(ws + 2 * XBYTES);
  float*    pots1 = pots0 + (size_t)4 * BB * NN;
  float*    fin   = pots1 + (size_t)4 * BB * NN;

  // geomloss epsilon schedule: [diam^p, exp(arange(p*ln diam, p*ln blur,
  //                             p*ln scaling)), blur^p]
  float eps_list[64];
  int   ne = 0;
  eps_list[ne++] = 4.0f;                               // DIAMETER^P
  {
    const double start = 2.0 * log(2.0);
    const double stop  = 2.0 * log(0.05);
    const double step  = 2.0 * log(0.9);
    for (int i = 0;; ++i) {
      const double v = start + (double)i * step;
      if (!(v > stop)) break;
      eps_list[ne++] = (float)exp(v);
    }
  }
  eps_list[ne++] = 0.05f * 0.05f;                      // BLUR^P  (ne == 38)

  // normalize + f32->f16
  sk_norm_kernel<<<BB * NN / 4, 128, 0, stream>>>(Ep, Xh, BB * NN);
  sk_norm_kernel<<<BB * NN / 4, 128, 0, stream>>>(Et, Yh, BB * NN);

  dim3 grid(NN / 64, BB, 4);  // 64 rows/block (4 waves x 16), B batches, 4 problems

  // init at eps0 (h = log weights only)
  sk_softmin_kernel<<<grid, 128, 0, stream>>>(Xh, Yh, pots0, pots0,
                                              eps_list[0], 0);
  // scan over full schedule with double-buffered potentials
  float* cur = pots0;
  float* nxt = pots1;
  for (int i = 0; i < ne; ++i) {
    sk_softmin_kernel<<<grid, 128, 0, stream>>>(Xh, Yh, cur, nxt,
                                                eps_list[i], 1);
    float* t = cur; cur = nxt; nxt = t;
  }
  // final extrapolation at the last eps (no averaging)
  sk_softmin_kernel<<<grid, 128, 0, stream>>>(Xh, Yh, cur, fin,
                                              eps_list[ne - 1], 2);
  // loss
  sk_loss_kernel<<<1, 256, 0, stream>>>(fin, out);
}
